// WQKV_56556129354122
// MI455X (gfx1250) — compile-verified, matrix operand order checked by reading, not run on hardware
//
#include <hip/hip_runtime.h>

#define T_SEQ   4096
#define CDIM    1024
#define BATCH   2
#define LOG_GAMMA (-0.0317486983145803f)   // ln(0.96875)

typedef _Float16 f16;
typedef __attribute__((ext_vector_type(16))) _Float16 v16h;
typedef __attribute__((ext_vector_type(8)))  float    v8f;

// ---------------------------------------------------------------------------
// WMMA fragment helpers (wave32, 16x16x32 f16 -> f32)
// 16-bit A/B layout (ISA 7.12.2): lane m/n = lane&15; hi-half of wave holds
// K groups {8..15, 24..31}; VGPR j<4 -> K = hi*8+2j, j>=4 -> K = 16+hi*8+2(j-4)
// A and B fragments have symmetric per-lane layouts, so a row-major [m][k]
// tile serves A and a row-major [n][k] tile serves B with the same loader,
// lowering to two ds_load_b128 per fragment.
// ---------------------------------------------------------------------------
__device__ __forceinline__ int frag_k(int j, int hi) {
  return (j < 4) ? (hi * 8 + 2 * j) : (16 + hi * 8 + 2 * (j - 4));
}

// element (k, m) = tile[(row0 + m)*stride + k0 + k]   (k-pairs contiguous)
__device__ __forceinline__ v16h load_frag_rowmajor(const f16* tile, int row0, int stride,
                                                   int k0, int lane) {
  int m = lane & 15, hi = lane >> 4;
  const f16* p = tile + (row0 + m) * stride + k0;
  v16h a;
#pragma unroll
  for (int j = 0; j < 8; ++j) {
    int k = frag_k(j, hi);
    a[2 * j]     = p[k];
    a[2 * j + 1] = p[k + 1];
  }
  return a;
}

__device__ __forceinline__ v8f wmma_f16(v16h a, v16h b, v8f c) {
  return __builtin_amdgcn_wmma_f32_16x16x32_f16(false, a, false, b, (short)0, c,
                                                false, false);
}

// CDNA5 async global->LDS copy (16B per lane), tracked by ASYNCcnt.
// lds_off is a byte offset from the wave's LDS base (attn kernel has no
// static LDS, so dynamic-LDS offset 0 == base).
__device__ __forceinline__ void async_copy_b128(unsigned int lds_off, const void* gptr) {
  asm volatile("global_load_async_to_lds_b128 %0, %1, off"
               :: "v"(lds_off), "v"((unsigned long long)gptr) : "memory");
}
__device__ __forceinline__ void wait_async0() {
  asm volatile("s_wait_asynccnt 0x0" ::: "memory");
}

// ---------------------------------------------------------------------------
// Kernel 1: projection GEMM  Y[M x 1024] = X[M x 1024] @ W[1024 x 1024]
// f32 in -> f16 out. Workgroup tile 64x128, 8 waves (4 row x 2 col), BK=32.
// Weight tile staged transposed [n][k] so B-fragments are b128 DS loads.
// ---------------------------------------------------------------------------
__global__ __launch_bounds__(256)
void proj_gemm_kernel(const float* __restrict__ X, const float* __restrict__ W,
                      f16* __restrict__ Y, int M) {
  __shared__ f16 As[64 * 40];    // [64 m][32 k] pad->40
  __shared__ f16 Bt[128 * 40];   // transposed [128 n][32 k] pad->40
  const int tid = threadIdx.x, lane = tid & 31, wave = tid >> 5;
  const int wr = wave >> 1, wc = wave & 1;
  const int m0 = blockIdx.y * 64, n0 = blockIdx.x * 128;
  (void)M;
  v8f acc[4] = {};

  for (int k0 = 0; k0 < CDIM; k0 += 32) {
    __syncthreads();
#pragma unroll
    for (int i = 0; i < 8; ++i) {             // A: 64x32 f32 -> f16
      int e = i * 256 + tid;
      int r = e >> 5, c = e & 31;
      As[r * 40 + c] = (f16)X[(size_t)(m0 + r) * CDIM + k0 + c];
    }
#pragma unroll
    for (int i = 0; i < 16; ++i) {            // B: 32x128 f32 -> f16, transpose in LDS
      int e = i * 256 + tid;
      int kr = e >> 7, n = e & 127;
      Bt[n * 40 + kr] = (f16)W[(size_t)(k0 + kr) * CDIM + n0 + n];
    }
    __syncthreads();
    v16h a = load_frag_rowmajor(As, wr * 16, 40, 0, lane);
#pragma unroll
    for (int nt = 0; nt < 4; ++nt) {
      v16h b = load_frag_rowmajor(Bt, wc * 64 + nt * 16, 40, 0, lane);
      acc[nt] = wmma_f16(a, b, acc[nt]);
    }
  }
  const int hi = lane >> 4, n = lane & 15;
#pragma unroll
  for (int nt = 0; nt < 4; ++nt)
#pragma unroll
    for (int i = 0; i < 8; ++i)
      Y[(size_t)(m0 + wr * 16 + i + 8 * hi) * CDIM + n0 + wc * 64 + nt * 16 + n] =
          (f16)acc[nt][i];
}

// ---------------------------------------------------------------------------
// Kernel 2: in-place RoPE on first 16 channels of each row (rot_dim = 16)
// ---------------------------------------------------------------------------
__global__ __launch_bounds__(256)
void rope_kernel(f16* __restrict__ X, int rows) {
  int idx = blockIdx.x * blockDim.x + threadIdx.x;
  if (idx >= rows * 8) return;
  int row = idx >> 3, p = idx & 7;
  int t = row & (T_SEQ - 1);
  float invf = __powf(10000.f, -(2.f * (float)p) / 16.f);
  float ang = (float)t * invf;
  float s, c;
  __sincosf(ang, &s, &c);
  size_t o = (size_t)row * CDIM + 2 * p;
  float x0 = (float)X[o], x1 = (float)X[o + 1];
  X[o]     = (f16)(x0 * c - x1 * s);
  X[o + 1] = (f16)(x1 * c + x0 * s);
}

// ---------------------------------------------------------------------------
// Kernel 3: fused causal-decay attention + group norm.
// Workgroup = 16 output rows x full 1024 cols, 8 waves own 128 cols each.
// 32-wide s-blocks: two Phase-A sub-steps (16 s each, K-split across waves,
// LDS reduce, decay*1/3 mask) fill Sfrag[16][32]; one full-K Phase B does
// O += S(16x32) @ V(32x128). K/Kr/Qr s-tiles staged via async global->LDS.
// Epilogue: GroupNorm (16 groups of 64) fused through LDS.
// ---------------------------------------------------------------------------
#define QT_OFF    0u                     // [16][1032] f16
#define KT_OFF    33024u                 // [16][1032] f16
#define KS_OFF    66048u                 // [16][1032] f16 (row-major s-tile)
#define KR_OFF    99072u
#define QR_OFF    132096u
#define VS_OFF    165120u                // [1024 chan][32 s] f16
#define SRED_OFF  230656u                // [8][16][16] f32
#define SFRAG_OFF 238848u                // [16][34] f16
#define ATTN_SMEM 239936u

__global__ __launch_bounds__(256)
void attn_gn_kernel(const f16* __restrict__ Qh, const f16* __restrict__ Kh,
                    const f16* __restrict__ Qrh, const f16* __restrict__ Krh,
                    const f16* __restrict__ Vh, const float* __restrict__ gnw,
                    const float* __restrict__ gnb, float* __restrict__ out) {
  extern __shared__ char smem_raw[];
  f16*   Qt    = (f16*)(smem_raw + QT_OFF);
  f16*   Kt    = (f16*)(smem_raw + KT_OFF);
  f16*   Ks    = (f16*)(smem_raw + KS_OFF);
  f16*   Kr    = (f16*)(smem_raw + KR_OFF);
  f16*   Qr    = (f16*)(smem_raw + QR_OFF);
  f16*   Vs    = (f16*)(smem_raw + VS_OFF);
  float* Sred  = (float*)(smem_raw + SRED_OFF);
  f16*   Sfrag = (f16*)(smem_raw + SFRAG_OFF);
  float* Ot    = (float*)smem_raw;       // epilogue alias [16][1040] f32

  const int tid = threadIdx.x, lane = tid & 31, wave = tid >> 5;
  const int t0 = blockIdx.x * 16, b = blockIdx.y;
  const size_t base = (size_t)b * T_SEQ * CDIM;
  const int wcol = wave * 128;

  // stage Q,K row tiles via async b128 copies (16B chunks, layout-preserving)
  for (int e = tid; e < 2048; e += 256) {          // 2048 chunks per tile
    int r = e >> 7, c = (e & 127) * 8;             // 128 chunks per 1024-half row
    unsigned int loff = (unsigned int)((r * 1032 + c) * 2);
    async_copy_b128(QT_OFF + loff, Qh + base + (size_t)(t0 + r) * CDIM + c);
    async_copy_b128(KT_OFF + loff, Kh + base + (size_t)(t0 + r) * CDIM + c);
  }

  v8f acc[8] = {};
  const int sEnd = t0 + 16;
  for (int s0 = 0; s0 < sEnd; s0 += 32) {
#pragma unroll 1
    for (int u = 0; u < 2; ++u) {
      const int srow = s0 + u * 16;       // max staged row = 4064+31 = 4095: in range
      __syncthreads();
      // stage K/Kr/Qr 16-row s-tiles row-major via async copies
      for (int e = tid; e < 2048; e += 256) {
        int s = e >> 7, c = (e & 127) * 8;
        unsigned int loff = (unsigned int)((s * 1032 + c) * 2);
        size_t gb = base + (size_t)(srow + s) * CDIM + c;
        size_t gr = (size_t)(srow + s) * CDIM + c;
        async_copy_b128(KS_OFF + loff, Kh + gb);
        async_copy_b128(KR_OFF + loff, Krh + gr);
        async_copy_b128(QR_OFF + loff, Qrh + gr);
      }
      if (u == 0) {
        // V needs a transpose (WMMA K-dim = s): stage all 32 s rows chan-major
        for (int e = tid; e < 8192; e += 256) {
          int s = e >> 8, c = (e & 255) * 4;
          uint2 vv = *(const uint2*)&Vh[base + (size_t)(s0 + s) * CDIM + c];
          union { uint2 un; f16 h[4]; } cv; cv.un = vv;
          Vs[(c + 0) * 32 + s] = cv.h[0];
          Vs[(c + 1) * 32 + s] = cv.h[1];
          Vs[(c + 2) * 32 + s] = cv.h[2];
          Vs[(c + 3) * 32 + s] = cv.h[3];
        }
      }
      wait_async0();
      __syncthreads();

      // prefetch next s-block into cache while matrix cores run
      if (u == 0 && s0 + 32 < sEnd) {
        int pr = tid >> 3, pc = (tid & 7) * 128;
        size_t pg = base + (size_t)(s0 + 32 + pr) * CDIM + pc;
        size_t pr2 = (size_t)(s0 + 32 + pr) * CDIM + pc;
        __builtin_prefetch(&Kh[pg], 0, 1);
        __builtin_prefetch(&Vh[pg], 0, 1);
        __builtin_prefetch(&Krh[pr2], 0, 1);
        __builtin_prefetch(&Qrh[pr2], 0, 1);
      }

      // Phase A: partial 16x16 scores, this wave reduces chans [wcol, wcol+128)
      v8f sacc = {};
#pragma unroll
      for (int kk = 0; kk < 128; kk += 32) {
        int kc = wcol + kk;
        v16h aQ  = load_frag_rowmajor(Qt, 0, 1032, kc, lane);
        v16h aK  = load_frag_rowmajor(Kt, 0, 1032, kc, lane);
        v16h bK  = load_frag_rowmajor(Ks, 0, 1032, kc, lane);
        v16h bKr = load_frag_rowmajor(Kr, 0, 1032, kc, lane);
        v16h bQr = load_frag_rowmajor(Qr, 0, 1032, kc, lane);
        sacc = wmma_f16(aQ, bK, sacc);
        sacc = wmma_f16(aQ, bKr, sacc);
        sacc = wmma_f16(aK, bQr, sacc);
      }
      {
        float* sr = Sred + wave * 256;
        int hi = lane >> 4, n = lane & 15;
#pragma unroll
        for (int i = 0; i < 8; ++i) sr[(i + 8 * hi) * 16 + n] = sacc[i];
      }
      __syncthreads();
      // cross-wave reduce + decay mask -> Sfrag columns [u*16, u*16+16)
      {
        int r = tid >> 4, n = tid & 15;
        float v = 0.f;
#pragma unroll
        for (int w = 0; w < 8; ++w) v += Sred[w * 256 + r * 16 + n];
        int t = t0 + r, s = srow + n;
        float wgt = (t >= s) ? __expf((float)(t - s) * LOG_GAMMA) * (1.f / 3.f) : 0.f;
        Sfrag[r * 34 + u * 16 + n] = (f16)(v * wgt);
      }
    }
    __syncthreads();

    // Phase B: O[16 x 128(wave)] += S(16x32) @ V(32x128), full K=32
    v16h aS = load_frag_rowmajor(Sfrag, 0, 34, 0, lane);
#pragma unroll
    for (int nt = 0; nt < 8; ++nt) {
      v16h bV = load_frag_rowmajor(Vs, wcol + nt * 16, 32, 0, lane);
      acc[nt] = wmma_f16(aS, bV, acc[nt]);
    }
  }

  // Epilogue: dump accumulators to LDS, fused GroupNorm (16 groups of 64)
  __syncthreads();
  {
    int hi = lane >> 4, n = lane & 15;
#pragma unroll
    for (int nt = 0; nt < 8; ++nt)
#pragma unroll
      for (int i = 0; i < 8; ++i)
        Ot[(i + 8 * hi) * 1040 + wcol + nt * 16 + n] = acc[nt][i];
  }
  __syncthreads();
  {
    int r = tid >> 4, g = tid & 15;
    const float4* row4 = (const float4*)(Ot + r * 1040 + g * 64);
    float4 buf[16];
    float mu = 0.f;
#pragma unroll
    for (int i = 0; i < 16; ++i) {
      buf[i] = row4[i];
      mu += buf[i].x + buf[i].y + buf[i].z + buf[i].w;
    }
    mu *= (1.f / 64.f);
    float var = 0.f;
#pragma unroll
    for (int i = 0; i < 16; ++i) {
      float dx = buf[i].x - mu, dy = buf[i].y - mu, dz = buf[i].z - mu, dw = buf[i].w - mu;
      var += dx * dx + dy * dy + dz * dz + dw * dw;
    }
    var *= (1.f / 64.f);
    float inv = rsqrtf(var + 1e-6f);
    size_t orow = ((size_t)b * T_SEQ + t0 + r) * CDIM + g * 64;
    const float4* w4 = (const float4*)(gnw + g * 64);
    const float4* b4 = (const float4*)(gnb + g * 64);
#pragma unroll
    for (int i = 0; i < 16; ++i) {
      float4 w = w4[i], bb = b4[i], o;
      o.x = (buf[i].x - mu) * inv * w.x + bb.x;
      o.y = (buf[i].y - mu) * inv * w.y + bb.y;
      o.z = (buf[i].z - mu) * inv * w.z + bb.z;
      o.w = (buf[i].w - mu) * inv * w.w + bb.w;
      *(float4*)&out[orow + i * 4] = o;
    }
  }
}

// ---------------------------------------------------------------------------
extern "C" void kernel_launch(void* const* d_in, const int* in_sizes, int n_in,
                              void* d_out, int out_size, void* d_ws, size_t ws_size,
                              hipStream_t stream) {
  (void)in_sizes; (void)n_in; (void)out_size; (void)ws_size;
  const float* Q_in    = (const float*)d_in[0];
  const float* rel_pos = (const float*)d_in[1];
  const float* W_Q     = (const float*)d_in[2];
  const float* W_K     = (const float*)d_in[3];
  const float* W_V     = (const float*)d_in[4];
  const float* W_Qr    = (const float*)d_in[5];
  const float* W_Kr    = (const float*)d_in[6];
  const float* gnw     = (const float*)d_in[7];
  const float* gnb     = (const float*)d_in[8];
  float* out = (float*)d_out;

  const size_t BT = (size_t)BATCH * T_SEQ;        // 8192
  f16* Qh  = (f16*)d_ws;                          // BT x C
  f16* Kh  = Qh  + BT * CDIM;
  f16* Vh  = Kh  + BT * CDIM;
  f16* Qrh = Vh  + BT * CDIM;                     // T x C
  f16* Krh = Qrh + (size_t)T_SEQ * CDIM;          // total 64 MB f16 workspace

  dim3 blk(256);
  // projections (f32 -> f16, WMMA)
  proj_gemm_kernel<<<dim3(8, 128), blk, 0, stream>>>(Q_in,    W_Q,  Qh,  8192);
  proj_gemm_kernel<<<dim3(8, 128), blk, 0, stream>>>(Q_in,    W_K,  Kh,  8192);
  proj_gemm_kernel<<<dim3(8, 128), blk, 0, stream>>>(Q_in,    W_V,  Vh,  8192);
  proj_gemm_kernel<<<dim3(8, 64),  blk, 0, stream>>>(rel_pos, W_Qr, Qrh, 4096);
  proj_gemm_kernel<<<dim3(8, 64),  blk, 0, stream>>>(rel_pos, W_Kr, Krh, 4096);
  // RoPE on Q, K (first 16 channels)
  rope_kernel<<<(8192 * 8 + 255) / 256, blk, 0, stream>>>(Qh, 8192);
  rope_kernel<<<(8192 * 8 + 255) / 256, blk, 0, stream>>>(Kh, 8192);
  // fused attention + group norm (235 KB dynamic LDS — needs CDNA5 320 KB WGP)
  hipFuncSetAttribute((const void*)attn_gn_kernel,
                      hipFuncAttributeMaxDynamicSharedMemorySize, (int)ATTN_SMEM);
  attn_gn_kernel<<<dim3(T_SEQ / 16, BATCH), blk, ATTN_SMEM, stream>>>(
      Qh, Kh, Qrh, Krh, Vh, gnw, gnb, out);
}
